// NeRF_14285061226818
// MI455X (gfx1250) — matrix operand order repaired and checked by reading, not compiled, not run on hardware
//
#include <hip/hip_runtime.h>

typedef __attribute__((ext_vector_type(16))) __bf16 v16bf;
typedef __attribute__((ext_vector_type(8)))  __bf16 v8bf;
typedef __attribute__((ext_vector_type(8)))  float  v8f;

#define STRIDE 336            // bf16 elems per LDS activation row (16B-aligned rows)
#define NWAVE 4               // waves per block
#define RPW 16                // rows per wave (WMMA M)
#define RPB (NWAVE*RPW)       // rows per block

// Workspace offsets (bf16 elements) for swizzled, padded weights.
constexpr int OFF_W0 = 0;                       // 64 x 256
constexpr int OFF_W1 = OFF_W0 + 64*256;         // 256 x 256
constexpr int OFF_W2 = OFF_W1 + 256*256;
constexpr int OFF_W3 = OFF_W2 + 256*256;
constexpr int OFF_W4 = OFF_W3 + 256*256;
constexpr int OFF_W5 = OFF_W4 + 256*256;        // 320 x 256 (row-permuted for skip concat)
constexpr int OFF_W6 = OFF_W5 + 320*256;
constexpr int OFF_W7 = OFF_W6 + 256*256;
constexpr int OFF_WF = OFF_W7 + 256*256;        // 256 x 256
constexpr int OFF_WA = OFF_WF + 256*256;        // 256 x 16 (N padded from 1)
constexpr int OFF_WV = OFF_WA + 256*16;         // 320 x 128
constexpr int OFF_WR = OFF_WV + 320*128;        // 128 x 16 (N padded from 3)
constexpr int WS_TOTAL = OFF_WR + 128*16;       // 604160 bf16 = ~1.2 MB

// ---------------------------------------------------------------------------
// Weight swizzle: fp32 row-major (K_real x N_real) -> bf16 B-fragment order.
// Dest layout: tile (kt,nt) at (nt*ktiles+kt)*512; within tile lane-major,
// lane l<16 holds column n=nt*16+l, K = kt*32 + {0..7, 16..23} (elements j ascending);
// lane l>=16 holds column n=nt*16+(l-16), K = kt*32 + {8..15, 24..31}.
// perm==1 applies the NeRF skip permutation (h first, pts last) to rows of W5.
// ---------------------------------------------------------------------------
__global__ void swizzle_w(const float* __restrict__ W, unsigned short* __restrict__ dst_,
                          int K_pad, int K_real, int N_out, int N_real, int perm) {
  __bf16* dst = reinterpret_cast<__bf16*>(dst_);
  int idx = blockIdx.x * blockDim.x + threadIdx.x;
  int total = K_pad * N_out;
  if (idx >= total) return;
  int j    = idx & 15;
  int l    = (idx >> 4) & 31;
  int tile = idx >> 9;
  int ktiles = K_pad >> 5;
  int kt = tile % ktiles;
  int nt = tile / ktiles;
  int k = kt * 32 + ((j & 8) << 1) + ((l & 16) >> 1) + (j & 7);
  int n = nt * 16 + (l & 15);
  float v = 0.0f;
  if (k < K_real && n < N_real) {
    int ks = perm ? ((k < 256) ? 63 + k : k - 256) : k;
    v = W[ks * N_real + n];
  }
  dst[idx] = (__bf16)v;
}

// ---------------------------------------------------------------------------
// Fragment helpers (wave32 WMMA 16x16x32 bf16).
// ---------------------------------------------------------------------------
template<int KT>
__device__ __forceinline__ void load_afrags(const __bf16* rowp, int lane, v16bf* afr) {
  // A 16x32 layout: lane<16 row M=lane holds K {0..7, 16..23}; lane>=16 same row, K {8..15, 24..31}
  const int koff = (lane & 16) ? 8 : 0;
#pragma unroll
  for (int kt = 0; kt < KT; ++kt) {
    v8bf lo = *(const v8bf*)(rowp + kt*32 + koff);        // ds_load_b128
    v8bf hi = *(const v8bf*)(rowp + kt*32 + koff + 16);   // ds_load_b128
    v16bf a;
#pragma unroll
    for (int i = 0; i < 8; ++i) { a[i] = lo[i]; a[i+8] = hi[i]; }
    afr[kt] = a;
  }
}

template<int KT>
__device__ __forceinline__ v8f tile_mm(const v16bf* afr, const __bf16* __restrict__ wbase, int lane) {
  v8f acc = {};
#pragma unroll
  for (int kt = 0; kt < KT; ++kt) {
    v16bf b = *(const v16bf*)(wbase + kt*512 + lane*16);  // 2x global_load_b128
    acc = __builtin_amdgcn_wmma_f32_16x16x32_bf16(false, afr[kt], false, b,
                                                  (short)0, acc, false, false);
  }
  return acc;
}

template<int KT, int NT, bool RELU>
__device__ __forceinline__ void dense_layer(const __bf16* rowp, __bf16* H, const __bf16* W,
                                            const float* __restrict__ bias, int lane) {
  v16bf afr[KT];
  load_afrags<KT>(rowp, lane, afr);        // hoisted: registers, safe vs. output overwrite
  const int mbase = (lane & 16) ? 8 : 0;   // C/D layout: VGPR r -> M = r + mbase, N = lane&15
  for (int nt = 0; nt < NT; ++nt) {
    v8f acc = tile_mm<KT>(afr, W + nt*KT*512, lane);
    int n = nt*16 + (lane & 15);
    float bb = bias[n];
#pragma unroll
    for (int r = 0; r < 8; ++r) {
      float v = acc[r] + bb;
      if (RELU) v = fmaxf(v, 0.0f);
      H[(mbase + r)*STRIDE + n] = (__bf16)v;
    }
  }
}

// ---------------------------------------------------------------------------
// Fused NeRF MLP: 4 waves/block, 16 rows/wave. Activations live in a private
// per-wave LDS panel; no cross-wave sharing -> no barriers (same-wave LDS is
// in-order). Panel cols: 0..255 = h, 256..318 = pts (then views), 319 = 0.
// ---------------------------------------------------------------------------
__global__ __launch_bounds__(128) void nerf_fused(
    const float* __restrict__ x,
    const float* __restrict__ b0, const float* __restrict__ b1,
    const float* __restrict__ b2, const float* __restrict__ b3,
    const float* __restrict__ b4, const float* __restrict__ b5,
    const float* __restrict__ b6, const float* __restrict__ b7,
    const float* __restrict__ bfp, const float* __restrict__ bap,
    const float* __restrict__ bvp, const float* __restrict__ brp,
    const unsigned short* __restrict__ wsw,
    float* __restrict__ out)
{
  __shared__ __align__(16) __bf16 hbuf[NWAVE][RPW][STRIDE];
  const __bf16* WS = reinterpret_cast<const __bf16*>(wsw);
  const int lane = threadIdx.x & 31;
  const int w    = threadIdx.x >> 5;
  const long long row0 = ((long long)blockIdx.x * NWAVE + w) * RPW;

  __bf16* H = &hbuf[w][0][0];
  const __bf16* rowp = &hbuf[w][lane & 15][0];

  // Stage pts (63 ch) into cols 0..62 (L0 input) and 256..318 (skip concat).
  for (int idx = lane; idx < RPW * 63; idx += 32) {
    int r = idx / 63, c = idx - r * 63;
    __bf16 v = (__bf16)x[(row0 + r) * 126 + c];
    H[r*STRIDE + c]       = v;
    H[r*STRIDE + 256 + c] = v;
  }
  {
    int r = lane & 15;
    H[r*STRIDE + 63]  = (__bf16)0.0f;   // K-pad for layer 0
    H[r*STRIDE + 319] = (__bf16)0.0f;   // K-pad for 320-wide layers
  }

  const float* bl[8] = {b0,b1,b2,b3,b4,b5,b6,b7};
  const int    wo[8] = {OFF_W0,OFF_W1,OFF_W2,OFF_W3,OFF_W4,OFF_W5,OFF_W6,OFF_W7};

  dense_layer<2,16,true>(rowp, H, WS + OFF_W0, b0, lane);                 // 63->256
  for (int i = 1; i <= 4; ++i)
    dense_layer<8,16,true>(rowp, H, WS + wo[i], bl[i], lane);             // 256->256
  dense_layer<10,16,true>(rowp, H, WS + OFF_W5, b5, lane);                // 319->256 (skip, perm folded)
  for (int i = 6; i <= 7; ++i)
    dense_layer<8,16,true>(rowp, H, WS + wo[i], bl[i], lane);             // 256->256

  // alpha (h@Wa+ba, col 0 only) + feature (h@Wf+bf, no relu) share A-frags of h.
  {
    v16bf afr[8];
    load_afrags<8>(rowp, lane, afr);
    const int mbase = (lane & 16) ? 8 : 0;
    v8f aacc = tile_mm<8>(afr, WS + OFF_WA, lane);
    if ((lane & 15) == 0) {
      float bb = bap[0];
#pragma unroll
      for (int r = 0; r < 8; ++r)
        out[(row0 + mbase + r) * 4 + 3] = aacc[r] + bb;                   // alpha
    }
    for (int nt = 0; nt < 16; ++nt) {
      v8f acc = tile_mm<8>(afr, WS + OFF_WF + nt*8*512, lane);
      int n = nt*16 + (lane & 15);
      float bb = bfp[n];
#pragma unroll
      for (int r = 0; r < 8; ++r)
        H[(mbase + r)*STRIDE + n] = (__bf16)(acc[r] + bb);                // feature (no relu)
    }
  }

  // Stage views into cols 256..318 -> hv = [feature(0..255), views(256..318)].
  for (int idx = lane; idx < RPW * 63; idx += 32) {
    int r = idx / 63, c = idx - r * 63;
    H[r*STRIDE + 256 + c] = (__bf16)x[(row0 + r) * 126 + 63 + c];
  }

  dense_layer<10,8,true>(rowp, H, WS + OFF_WV, bvp, lane);                // 319->128

  // rgb = hv@Wr + br (cols 0..2 valid)
  {
    v16bf afr[4];
    load_afrags<4>(rowp, lane, afr);
    v8f racc = tile_mm<4>(afr, WS + OFF_WR, lane);
    int n = lane & 15;
    if (n < 3) {
      float bb = brp[n];
      const int mbase = (lane & 16) ? 8 : 0;
#pragma unroll
      for (int r = 0; r < 8; ++r)
        out[(row0 + mbase + r) * 4 + n] = racc[r] + bb;
    }
  }
}

// ---------------------------------------------------------------------------
extern "C" void kernel_launch(void* const* d_in, const int* in_sizes, int n_in,
                              void* d_out, int out_size, void* d_ws, size_t ws_size,
                              hipStream_t stream) {
  const float* x = (const float*)d_in[0];
  const float* W[8]; const float* b[8];
  for (int i = 0; i < 8; ++i) { W[i] = (const float*)d_in[1 + 2*i]; b[i] = (const float*)d_in[2 + 2*i]; }
  const float* Wf = (const float*)d_in[17]; const float* bf_ = (const float*)d_in[18];
  const float* Wa = (const float*)d_in[19]; const float* ba_ = (const float*)d_in[20];
  const float* Wv = (const float*)d_in[21]; const float* bv_ = (const float*)d_in[22];
  const float* Wr = (const float*)d_in[23]; const float* br_ = (const float*)d_in[24];
  unsigned short* ws = (unsigned short*)d_ws;

  auto sw = [&](const float* Wp, int off, int Kp, int Kr, int No, int Nr, int perm) {
    int total = Kp * No;
    swizzle_w<<<(total + 255) / 256, 256, 0, stream>>>(Wp, ws + off, Kp, Kr, No, Nr, perm);
  };
  sw(W[0], OFF_W0,  64,  63, 256, 256, 0);
  sw(W[1], OFF_W1, 256, 256, 256, 256, 0);
  sw(W[2], OFF_W2, 256, 256, 256, 256, 0);
  sw(W[3], OFF_W3, 256, 256, 256, 256, 0);
  sw(W[4], OFF_W4, 256, 256, 256, 256, 0);
  sw(W[5], OFF_W5, 320, 319, 256, 256, 1);   // skip permutation folded into rows
  sw(W[6], OFF_W6, 256, 256, 256, 256, 0);
  sw(W[7], OFF_W7, 256, 256, 256, 256, 0);
  sw(Wf,   OFF_WF, 256, 256, 256, 256, 0);
  sw(Wa,   OFF_WA, 256, 256,  16,   1, 0);
  sw(Wv,   OFF_WV, 320, 319, 128, 128, 0);
  sw(Wr,   OFF_WR, 128, 128,  16,   3, 0);

  int nrows = in_sizes[0] / 126;             // 131072
  int grid  = nrows / RPB;                   // 2048 blocks, 64 rows each
  nerf_fused<<<grid, 128, 0, stream>>>(x,
      b[0], b[1], b[2], b[3], b[4], b[5], b[6], b[7],
      bf_, ba_, bv_, br_, ws, (float*)d_out);
}
